// OrthogonalBivectorBlock_37082747633939
// MI455X (gfx1250) — matrix operand order repaired and checked by reading, not compiled, not run on hardware
//
#include <hip/hip_runtime.h>
#include <math.h>

#define BB 2
#define LL 2048
#define DD 256
#define NTOK (BB * LL)
#define KDIM 256
#define CH 16                 // scan chunks along L
#define CL (LL / CH)          // 128 tokens per chunk
#define PI_F 3.14159265358979323846f

typedef __attribute__((ext_vector_type(16))) _Float16 v16h;
typedef __attribute__((ext_vector_type(8)))  float    v8f;

__device__ __forceinline__ float gelu_erf(float x) {
    return 0.5f * x * (1.0f + erff(x * 0.7071067811865476f));
}
__device__ __forceinline__ float sigmoidf_(float x) {
    return 1.0f / (1.0f + expf(-x));
}

// ---------------------------------------------------------------------------
// Async staging of one token's shared record into LDS (144 floats = 36 x 16B):
//   sh[0..111]  = PH[tok] (packed phasor record)
//   sh[112..143]= POSB[l] (positional phasors)
// Wave 0 issues GLOBAL_LOAD_ASYNC_TO_LDS_B128 per lane (ASYNCcnt-tracked),
// waits on its own async counter; the caller's __syncthreads releases readers.
// LDS operand = low 32 bits of the flat shared pointer (ISA 10.2: LDS aperture
// truncates flat address to addr[31:0]).
// ---------------------------------------------------------------------------
__device__ __forceinline__ void stage_async(int tid, unsigned lds_base,
                                            const float* __restrict__ ph_tok,
                                            const float* __restrict__ pos_l)
{
    if (tid < 32) {
        const float* src = (tid < 28) ? (ph_tok + tid * 4) : (pos_l + (tid - 28) * 4);
        unsigned dst = lds_base + tid * 16;
        asm volatile("global_load_async_to_lds_b128 %0, %1, off"
                     :: "v"(dst), "v"((unsigned long long)(size_t)src)
                     : "memory");
        if (tid < 4) {
            const float* src2 = pos_l + 16 + tid * 4;
            unsigned dst2 = lds_base + 512 + tid * 16;
            asm volatile("global_load_async_to_lds_b128 %0, %1, off"
                         :: "v"(dst2), "v"((unsigned long long)(size_t)src2)
                         : "memory");
        }
#if __has_builtin(__builtin_amdgcn_s_wait_asynccnt)
        __builtin_amdgcn_s_wait_asynccnt(0);
#else
        asm volatile("s_wait_asynccnt 0" ::: "memory");
#endif
    }
}

// ---------------------------------------------------------------------------
// C[M,N] = act(A[M,256] @ W[256,N] + bias[N]) via v_wmma_f32_16x16x32_f16.
// One wave per 16x16 tile; K fully unrolled -> 8 static WMMA.
// Fragment layouts per CDNA5 ISA 7.12.2:
//   A (16x32 f16): lane&15 = M row; 16 halves cover K:
//     k = (v>>2)*16 + (v&3)*2 + (h&1) + (lane>=16 ? 8 : 0), v = h>>1
//   B (32x16 f16): lane&15 = N column; 16 halves cover K in the same order as A
//   C/D (16x16 f32): VGPR j -> M = j + (lane>=16 ? 8 : 0), N = lane&15
// ACT: 0=none, 1=gelu(erf), 2=tanh()*pi
// ---------------------------------------------------------------------------
template <int ACT>
__global__ void wmma_gemm(const float* __restrict__ A, const float* __restrict__ W,
                          const float* __restrict__ bias, float* __restrict__ C,
                          int M, int N)
{
    int wid  = (blockIdx.x * blockDim.x + threadIdx.x) >> 5;
    int lane = threadIdx.x & 31;
    int tilesN = N >> 4;
    int tm = wid / tilesN;
    int tn = wid % tilesN;
    if (tm >= (M >> 4)) return;

    int hi = lane >> 4;
    int lo = lane & 15;

    const float* Arow = A + (size_t)(tm * 16 + lo) * KDIM;
    v8f acc = {};

#pragma unroll
    for (int k0 = 0; k0 < KDIM; k0 += 32) {
        v16h a, b;
#pragma unroll
        for (int h = 0; h < 16; ++h) {
            int v  = h >> 1;
            int kk = ((v >> 2) << 4) + ((v & 3) << 1) + (h & 1) + (hi << 3);
            a[h] = (_Float16)Arow[k0 + kk];
        }
        const float* Wcol = W + (size_t)k0 * N + tn * 16 + lo;
#pragma unroll
        for (int h = 0; h < 16; ++h) {
            int v  = h >> 1;
            int kk = ((v >> 2) << 4) + ((v & 3) << 1) + (h & 1) + (hi << 3);
            b[h] = (_Float16)Wcol[(size_t)kk * N];
        }
        acc = __builtin_amdgcn_wmma_f32_16x16x32_f16(
            false, a, false, b, (short)0, acc, false, false);
    }

#pragma unroll
    for (int j = 0; j < 8; ++j) {
        int m = tm * 16 + j + hi * 8;
        int n = tn * 16 + lo;
        float v = acc[j] + bias[n];
        if (ACT == 1)      v = gelu_erf(v);
        else if (ACT == 2) v = tanhf(v) * PI_F;
        C[(size_t)m * N + n] = v;
    }
}

// gate = sigmoid(G1[t,:] . gate_w2 + gate_b2), K=128, N=1
__global__ void gate_out_kernel(const float* __restrict__ G1,
                                const float* __restrict__ w, const float* __restrict__ b,
                                float* __restrict__ gate)
{
    int t = blockIdx.x * blockDim.x + threadIdx.x;
    if (t >= NTOK) return;
    const float* g = G1 + (size_t)t * 128;
    float s = b[0];
#pragma unroll 8
    for (int i = 0; i < 128; ++i) s += g[i] * w[i];
    gate[t] = sigmoidf_(s);
}

// ---------------------------------------------------------------------------
// Packed per-token phasor record PH (112 floats):
//   [0..15]  key cos     [16..31] key sin
//   [32..47] query cos   [48..63] query sin
//   [64..79] lke cos     [80..95] lke sin
//   [96..99] joint-key re  [100..103] joint-key im
//   [104..107] joint-query re  [108..111] joint-query im
// POSB (per position l, 32 floats): [0..15] cos, [16..31] sin
// ---------------------------------------------------------------------------
__global__ void phasor_kernel(const float* __restrict__ KANG, const float* __restrict__ QANG,
                              const float* __restrict__ LANG,
                              float* __restrict__ PH,
                              const float* __restrict__ pos_freqs, float* __restrict__ POSB)
{
    int t = blockIdx.x * blockDim.x + threadIdx.x;
    if (t >= NTOK) return;
    float kc[16], ks[16], qc[16], qs[16];
    size_t base = (size_t)t * 16;
    float* ph = PH + (size_t)t * 112;
#pragma unroll
    for (int p = 0; p < 16; ++p) {
        float s, c;
        sincosf(KANG[base + p], &s, &c); kc[p] = c; ks[p] = s; ph[p] = c; ph[16 + p] = s;
        sincosf(QANG[base + p], &s, &c); qc[p] = c; qs[p] = s; ph[32 + p] = c; ph[48 + p] = s;
        sincosf(LANG[base + p], &s, &c); ph[64 + p] = c; ph[80 + p] = s;
    }
#pragma unroll
    for (int pp = 0; pp < 4; ++pp) {
        float jr = 1.f, ji = 0.f, qr = 1.f, qi = 0.f;
#pragma unroll
        for (int s = 0; s < 4; ++s) {
            int p = s * 4 + pp;
            float nr = jr * kc[p] - ji * ks[p];
            ji = jr * ks[p] + ji * kc[p]; jr = nr;
            nr = qr * qc[p] - qi * qs[p];
            qi = qr * qs[p] + qi * qc[p]; qr = nr;
        }
        ph[96 + pp] = jr; ph[100 + pp] = ji;
        ph[104 + pp] = qr; ph[108 + pp] = qi;
    }
    if (t < LL) {
#pragma unroll
        for (int p = 0; p < 16; ++p) {
            float s, c;
            sincosf((float)t * pos_freqs[p] * (2.0f * PI_F), &s, &c);
            POSB[t * 32 + p] = c; POSB[t * 32 + 16 + p] = s;
        }
    }
}

// |shifted cumsum of joint_key| per (b, plane). |joint_query|==1 so the
// resonance magnitude is just |key_memory|.
__global__ void surprise_scan_kernel(const float* __restrict__ PH, float* __restrict__ KMMAG)
{
    int id = threadIdx.x;
    if (id >= 8) return;
    int b = id >> 2, pp = id & 3;
    float kr = 0.f, ki = 0.f;
    for (int t = 0; t < LL; ++t) {
        int tok = b * LL + t;
        KMMAG[tok * 4 + pp] = sqrtf(kr * kr + ki * ki);
        kr += PH[(size_t)tok * 112 + 96 + pp];
        ki += PH[(size_t)tok * 112 + 100 + pp];
    }
}

__global__ void write_gate_kernel(const float* __restrict__ KMMAG,
                                  const float* __restrict__ p_ss, const float* __restrict__ p_sb,
                                  const float* __restrict__ p_rs, const float* __restrict__ p_rt,
                                  float* __restrict__ WG)
{
    int tok = blockIdx.x * blockDim.x + threadIdx.x;
    if (tok >= NTOK) return;
    int l = tok % LL;
    float res = 0.25f * (KMMAG[tok * 4 + 0] + KMMAG[tok * 4 + 1] +
                         KMMAG[tok * 4 + 2] + KMMAG[tok * 4 + 3]);
    float scale = fminf(fmaxf(p_rs[0], 1.0f), 20.0f);
    float thr   = fminf(fmaxf(p_rt[0], 0.1f), 0.9f);
    float pos   = fmaxf((float)l, 1.0f);
    float nr = res * rsqrtf(pos);
    float surprise = 0.5f * (1.0f - tanhf(scale * (nr - thr)));
    WG[tok] = sigmoidf_(p_ss[0] * (surprise - 0.5f) + p_sb[0]);
}

__global__ void vg_kernel(const float* __restrict__ V, const float* __restrict__ WG,
                          float* __restrict__ Vg)
{
    int i = blockIdx.x * blockDim.x + threadIdx.x;
    if (i >= NTOK * DD) return;
    Vg[i] = V[i] * WG[i / DD];
}

// ---------------------------------------------------------------------------
// Chunked parallel scan. 36 complex accumulators per (b,d):
// components: [0..15] akr, [16..31] aki, [32..35] cjr, [36..39] cji,
//             [40..55] ppr, [56..71] ppi        (72 floats)
// ---------------------------------------------------------------------------

// pass 1: per-(b,d,chunk) partial bound sums
__global__ void scan_partial_kernel(const float* __restrict__ V, const float* __restrict__ Vg,
                                    const float* __restrict__ PH, const float* __restrict__ POSB,
                                    float* __restrict__ PART)
{
    __shared__ __align__(16) float sh[144];
    int b  = blockIdx.x / CH;
    int ch = blockIdx.x % CH;
    int d  = threadIdx.x;
    unsigned lds_base = (unsigned)(size_t)&sh[0];

    float akr[16] = {0.f}, aki[16] = {0.f};
    float cjr[4]  = {0.f}, cji[4]  = {0.f};
    float ppr[16] = {0.f}, ppi[16] = {0.f};

    int l0 = ch * CL;
    for (int l = l0; l < l0 + CL; ++l) {
        int tok = b * LL + l;
        __syncthreads();                     // readers of previous token done
        stage_async(d, lds_base, PH + (size_t)tok * 112, POSB + (size_t)l * 32);
        __syncthreads();                     // staged data visible
        float vg = Vg[(size_t)tok * DD + d];
        float vv = V[(size_t)tok * DD + d];
        __builtin_prefetch(Vg + (size_t)(tok + 1) * DD + d, 0, 1);
        __builtin_prefetch(V  + (size_t)(tok + 1) * DD + d, 0, 1);
#pragma unroll
        for (int p = 0; p < 16; ++p) { akr[p] += sh[p] * vg; aki[p] += sh[16 + p] * vg; }
#pragma unroll
        for (int p = 0; p < 4; ++p)  { cjr[p] += sh[96 + p] * vg; cji[p] += sh[100 + p] * vg; }
#pragma unroll
        for (int p = 0; p < 16; ++p) { ppr[p] += sh[112 + p] * vv; ppi[p] += sh[128 + p] * vv; }
    }

    float* out = PART + ((size_t)(b * DD + d) * CH + ch) * 72;
#pragma unroll
    for (int p = 0; p < 16; ++p) {
        out[p] = akr[p]; out[16 + p] = aki[p];
        out[40 + p] = ppr[p]; out[56 + p] = ppi[p];
    }
#pragma unroll
    for (int p = 0; p < 4; ++p) { out[32 + p] = cjr[p]; out[36 + p] = cji[p]; }
}

// pass 2: exclusive prefix over chunks, independently per (b,d,component)
__global__ void scan_prefix_kernel(float* __restrict__ PART)
{
    int idx = blockIdx.x * blockDim.x + threadIdx.x;
    if (idx >= BB * DD * 72) return;
    int comp = idx % 72;
    int bd   = idx / 72;
    float* base = PART + (size_t)bd * CH * 72 + comp;
    float running = 0.f;
    for (int ch = 0; ch < CH; ++ch) {
        float tmp = base[ch * 72];
        base[ch * 72] = running;
        running += tmp;
    }
}

// pass 3: rescan each chunk from its prefix offset and emit outputs
__global__ void scan_final_kernel(const float* __restrict__ V, const float* __restrict__ Vg,
                                  const float* __restrict__ PH, const float* __restrict__ POSB,
                                  const float* __restrict__ PART, const float* __restrict__ GATE,
                                  const float* __restrict__ set_weights,
                                  const float* __restrict__ p_posw, const float* __restrict__ p_ltmw,
                                  const float* __restrict__ ltm_mem, const float* __restrict__ p_ltm_count,
                                  float* __restrict__ TOTAL)
{
    __shared__ __align__(16) float sh[144];
    int b  = blockIdx.x / CH;
    int ch = blockIdx.x % CH;
    int d  = threadIdx.x;
    unsigned lds_base = (unsigned)(size_t)&sh[0];

    // softmax over 4 set weights
    float wsm[4];
    float mx = fmaxf(fmaxf(set_weights[0], set_weights[1]),
                     fmaxf(set_weights[2], set_weights[3]));
    float den = 0.f;
#pragma unroll
    for (int s = 0; s < 4; ++s) { wsm[s] = expf(set_weights[s] - mx); den += wsm[s]; }
#pragma unroll
    for (int s = 0; s < 4; ++s) wsm[s] /= den;

    float swpos = sigmoidf_(p_posw[0]);
    float swltm = sigmoidf_(p_ltmw[0]);
    float pnorm_inv = rsqrtf(fmaxf(p_ltm_count[0], 1.0f) * 16.0f);

    float Mre[16], Mim[16];
#pragma unroll
    for (int p = 0; p < 16; ++p) {
        Mre[p] = ltm_mem[2 * (p * DD + d) + 0];
        Mim[p] = ltm_mem[2 * (p * DD + d) + 1];
    }

    const float* off = PART + ((size_t)(b * DD + d) * CH + ch) * 72;
    float akr[16], aki[16], cjr[4], cji[4], ppr[16], ppi[16];
#pragma unroll
    for (int p = 0; p < 16; ++p) {
        akr[p] = off[p]; aki[p] = off[16 + p];
        ppr[p] = off[40 + p]; ppi[p] = off[56 + p];
    }
#pragma unroll
    for (int p = 0; p < 4; ++p) { cjr[p] = off[32 + p]; cji[p] = off[36 + p]; }

    int l0 = ch * CL;
    for (int l = l0; l < l0 + CL; ++l) {
        int tok = b * LL + l;
        __syncthreads();
        stage_async(d, lds_base, PH + (size_t)tok * 112, POSB + (size_t)l * 32);
        __syncthreads();
        float vg = Vg[(size_t)tok * DD + d];
        float vv = V[(size_t)tok * DD + d];
        __builtin_prefetch(Vg + (size_t)(tok + 1) * DD + d, 0, 1);
        __builtin_prefetch(V  + (size_t)(tok + 1) * DD + d, 0, 1);

        float setsum[4] = {0.f, 0.f, 0.f, 0.f};
#pragma unroll
        for (int p = 0; p < 16; ++p) {
            akr[p] += sh[p] * vg;       // key cos
            aki[p] += sh[16 + p] * vg;  // key sin
            setsum[p >> 2] += akr[p] * sh[32 + p] + aki[p] * sh[48 + p];  // Re(mem*conj(q))
        }
        float tot = wsm[0] * setsum[0] + wsm[1] * setsum[1] +
                    wsm[2] * setsum[2] + wsm[3] * setsum[3];

        float cross = 0.f;
#pragma unroll
        for (int p = 0; p < 4; ++p) {
            cjr[p] += sh[96 + p] * vg;
            cji[p] += sh[100 + p] * vg;
            cross += cjr[p] * sh[104 + p] + cji[p] * sh[108 + p];
        }
        tot = (tot + cross) * 0.2f;   // / (N_SETS + 1)

        float rp = 0.f;
#pragma unroll
        for (int p = 0; p < 16; ++p) {
            ppr[p] += sh[112 + p] * vv;
            ppi[p] += sh[128 + p] * vv;
            rp += ppr[p] * sh[112 + p] + ppi[p] * sh[128 + p];
        }

        float pers = 0.f;
#pragma unroll
        for (int p = 0; p < 16; ++p)
            pers += Mre[p] * sh[64 + p] + Mim[p] * sh[80 + p];

        float g = GATE[tok];
        TOTAL[(size_t)tok * DD + d] =
            g * tot + (1.0f - g) * (swpos * rp) + swltm * pers * pnorm_inv;
    }
}

__global__ void layernorm_kernel(const float* __restrict__ TOTAL,
                                 const float* __restrict__ gam, const float* __restrict__ bet,
                                 float* __restrict__ H)
{
    __shared__ float sm[DD];
    int tok = blockIdx.x;
    int d = threadIdx.x;
    float x = TOTAL[(size_t)tok * DD + d];
    sm[d] = x;
    __syncthreads();
    for (int s = DD / 2; s > 0; s >>= 1) {
        if (d < s) sm[d] += sm[d + s];
        __syncthreads();
    }
    float mu = sm[0] * (1.0f / DD);
    __syncthreads();
    float dev = x - mu;
    sm[d] = dev * dev;
    __syncthreads();
    for (int s = DD / 2; s > 0; s >>= 1) {
        if (d < s) sm[d] += sm[d + s];
        __syncthreads();
    }
    float var = sm[0] * (1.0f / DD);
    H[(size_t)tok * DD + d] = dev * rsqrtf(var + 1e-5f) * gam[d] + bet[d];
}

// ---------------------------------------------------------------------------

extern "C" void kernel_launch(void* const* d_in, const int* in_sizes, int n_in,
                              void* d_out, int out_size, void* d_ws, size_t ws_size,
                              hipStream_t stream)
{
    (void)in_sizes; (void)n_in; (void)out_size; (void)ws_size;

    const float* x        = (const float*)d_in[0];
    const float* ke_w1    = (const float*)d_in[1];
    const float* ke_b1    = (const float*)d_in[2];
    const float* ke_w2    = (const float*)d_in[3];
    const float* ke_b2    = (const float*)d_in[4];
    const float* qe_w1    = (const float*)d_in[5];
    const float* qe_b1    = (const float*)d_in[6];
    const float* qe_w2    = (const float*)d_in[7];
    const float* qe_b2    = (const float*)d_in[8];
    const float* v_w      = (const float*)d_in[9];
    const float* v_b      = (const float*)d_in[10];
    const float* out_ln_g = (const float*)d_in[11];
    const float* out_ln_b = (const float*)d_in[12];
    const float* out_w    = (const float*)d_in[13];
    const float* out_b    = (const float*)d_in[14];
    const float* set_w    = (const float*)d_in[15];
    const float* pos_wt   = (const float*)d_in[16];
    const float* gate_w1  = (const float*)d_in[17];
    const float* gate_b1  = (const float*)d_in[18];
    const float* gate_w2  = (const float*)d_in[19];
    const float* gate_b2  = (const float*)d_in[20];
    const float* lke_w1   = (const float*)d_in[21];
    const float* lke_b1   = (const float*)d_in[22];
    const float* lke_w2   = (const float*)d_in[23];
    const float* lke_b2   = (const float*)d_in[24];
    const float* sur_sc   = (const float*)d_in[25];
    const float* sur_bi   = (const float*)d_in[26];
    const float* res_sc   = (const float*)d_in[27];
    const float* res_th   = (const float*)d_in[28];
    const float* ltm_wt   = (const float*)d_in[29];
    const float* pos_fr   = (const float*)d_in[30];
    const float* ltm_mem  = (const float*)d_in[31];  // complex64 interleaved
    const float* ltm_cnt  = (const float*)d_in[32];

    float* ws = (float*)d_ws;
    // workspace layout (float offsets)
    float* Hbuf = ws + 0;          // 1,048,576 (reused hidden buffer)
    float* Vbuf = ws + 1048576;    // 1,048,576
    float* Vg   = ws + 2097152;    // 1,048,576
    float* TOT  = ws + 3145728;    // 1,048,576
    float* HLN  = ws + 4194304;    // 1,048,576
    float* KANG = ws + 5242880;    // 65,536
    float* QANG = ws + 5308416;    // 65,536
    float* LANG = ws + 5373952;    // 65,536
    float* PH   = ws + 5439488;    // 4096*112 = 458,752
    float* POSB = ws + 5898240;    // 2048*32  = 65,536
    float* PART = ws + 5963776;    // 2*256*16*72 = 589,824
    float* KMM  = ws + 6553600;    // 16,384
    float* WG   = ws + 6569984;    // 4,096
    float* GATE = ws + 6574080;    // 4,096
    // end: 6,578,176 floats (~26.3 MB)

    auto gemm = [&](const float* A, const float* W, const float* bias, float* C,
                    int M, int N, int act) {
        int waves  = (M / 16) * (N / 16);
        int blocks = (waves + 7) / 8;   // 8 waves (256 threads) per block
        if (act == 0)      wmma_gemm<0><<<dim3(blocks), dim3(256), 0, stream>>>(A, W, bias, C, M, N);
        else if (act == 1) wmma_gemm<1><<<dim3(blocks), dim3(256), 0, stream>>>(A, W, bias, C, M, N);
        else               wmma_gemm<2><<<dim3(blocks), dim3(256), 0, stream>>>(A, W, bias, C, M, N);
    };

    // phase 1: token-parallel GEMMs on the matrix pipes (all K = 256)
    gemm(x,    ke_w1,   ke_b1,   Hbuf, NTOK, DD,  1);
    gemm(Hbuf, ke_w2,   ke_b2,   KANG, NTOK, 16,  2);
    gemm(x,    qe_w1,   qe_b1,   Hbuf, NTOK, DD,  1);
    gemm(Hbuf, qe_w2,   qe_b2,   QANG, NTOK, 16,  2);
    gemm(x,    lke_w1,  lke_b1,  Hbuf, NTOK, DD,  1);
    gemm(Hbuf, lke_w2,  lke_b2,  LANG, NTOK, 16,  2);
    gemm(x,    v_w,     v_b,     Vbuf, NTOK, DD,  0);
    gemm(x,    gate_w1, gate_b1, Hbuf, NTOK, 128, 1);
    gate_out_kernel<<<dim3((NTOK + 255) / 256), dim3(256), 0, stream>>>(
        Hbuf, gate_w2, gate_b2, GATE);

    // phase 2: phasors -> packed records + positional table
    phasor_kernel<<<dim3(NTOK / 256), dim3(256), 0, stream>>>(
        KANG, QANG, LANG, PH, pos_fr, POSB);

    // phase 3: surprise gate (|joint_query|==1 -> only |cumsum(joint_key)|)
    surprise_scan_kernel<<<dim3(1), dim3(8), 0, stream>>>(PH, KMM);
    write_gate_kernel<<<dim3(NTOK / 256), dim3(256), 0, stream>>>(
        KMM, sur_sc, sur_bi, res_sc, res_th, WG);
    vg_kernel<<<dim3(NTOK * DD / 256), dim3(256), 0, stream>>>(Vbuf, WG, Vg);

    // phase 4: chunked parallel scan (partial -> prefix -> final)
    scan_partial_kernel<<<dim3(BB * CH), dim3(DD), 0, stream>>>(Vbuf, Vg, PH, POSB, PART);
    scan_prefix_kernel<<<dim3((BB * DD * 72 + 255) / 256), dim3(256), 0, stream>>>(PART);
    scan_final_kernel<<<dim3(BB * CH), dim3(DD), 0, stream>>>(
        Vbuf, Vg, PH, POSB, PART, GATE, set_w, pos_wt, ltm_wt, ltm_mem, ltm_cnt, TOT);

    // phase 5: LayerNorm + out projection (WMMA) -> d_out
    layernorm_kernel<<<dim3(NTOK), dim3(DD), 0, stream>>>(TOT, out_ln_g, out_ln_b, HLN);
    gemm(HLN, out_w, out_b, (float*)d_out, NTOK, DD, 0);
}
